// Mamba2Block_31215822307424
// MI455X (gfx1250) — matrix operand order repaired and checked by reading, not compiled, not run on hardware
//
#include <hip/hip_runtime.h>
#include <hip/hip_bf16.h>

// ---------------- problem constants ----------------
static constexpr int D_MODEL = 1024;
static constexpr int D_STATE = 64;
static constexpr int D_CONV  = 4;
static constexpr int N_HEADS = 8;
static constexpr int D_INNER = 2048;           // EXPAND * D_MODEL
static constexpr int D_HEAD  = 256;            // D_INNER / N_HEADS
static constexpr int BATCH   = 2;
static constexpr int SEQ     = 2048;
static constexpr int MROWS   = BATCH * SEQ;    // 4096 token rows
static constexpr int N_XZ    = 2 * D_INNER;    // 4096
static constexpr int N_BCDT  = N_HEADS * D_STATE * 2 + N_HEADS; // 512+512+8 = 1032

typedef __attribute__((ext_vector_type(16))) __bf16 v16bf;
typedef __attribute__((ext_vector_type(8)))  float  v8f;
typedef __attribute__((ext_vector_type(4)))  unsigned int v4u;
typedef __attribute__((ext_vector_type(4)))  int    v4i;
typedef __attribute__((ext_vector_type(8)))  int    v8i;

// ---------------- helpers ----------------
__device__ __forceinline__ float silu_f(float v) {
    return v / (1.0f + __expf(-v));
}

__device__ __forceinline__ v16bf load_frag16(const __bf16* p0, const __bf16* p1) {
    v4u tmp[2];
    tmp[0] = *(const v4u*)p0;
    tmp[1] = *(const v4u*)p1;
    v16bf r;
    __builtin_memcpy(&r, tmp, 32);
    return r;
}

__device__ __forceinline__ unsigned lds_off_of(const void* p) {
    // generic shared pointer: addr[31:0] is the LDS byte offset (aperture mapping)
    return (unsigned)(unsigned long long)p;
}

// ---- Tensor Data Mover: DMA a 2D tile [tile_rows x tile_cols elems] into LDS ----
// D# per CDNA5 ISA. Rows beyond rows_valid read as zero.
// dsize_code: 0=1B, 1=2B (bf16), 2=4B (f32), 3=8B.
__device__ __forceinline__ void tdm_load_2d(
    unsigned lds_byte_addr, const void* gptr,
    unsigned rows_valid, unsigned tile_rows,
    unsigned tile_cols, unsigned row_stride_elems, unsigned dsize_code) {
    unsigned long long ga = (unsigned long long)gptr;

    v4u g0;
    g0[0] = 1u;                                            // count=1, user descriptor
    g0[1] = lds_byte_addr;                                 // lds_addr [63:32]
    g0[2] = (unsigned)(ga & 0xffffffffu);                  // global_addr[31:0]
    g0[3] = (unsigned)((ga >> 32) & 0x01ffffffu) | (2u << 30); // addr[56:32] | type=2

    v8i g1;
    g1[0] = (int)(dsize_code << 16);                       // data_size; mask/pad=0
    g1[1] = (int)((tile_cols & 0xffffu) << 16);            // tensor_dim0[15:0] (=tile width)
    g1[2] = (int)((rows_valid & 0xffffu) << 16);           // tensor_dim0 hi=0 | tensor_dim1 lo
    g1[3] = (int)(((rows_valid >> 16) & 0xffffu) |
                  ((tile_cols & 0xffffu) << 16));          // tensor_dim1 hi | tile_dim0
    g1[4] = (int)(tile_rows & 0xffffu);                    // tile_dim1 | tile_dim2=0
    g1[5] = (int)row_stride_elems;                         // tensor_dim0_stride[31:0]
    g1[6] = 0;                                             // stride0 hi | stride1 lo
    g1[7] = 0;

    v4i z4 = {0, 0, 0, 0};
#if __clang_major__ >= 23
    v8i z8 = {0, 0, 0, 0, 0, 0, 0, 0};
    __builtin_amdgcn_tensor_load_to_lds(g0, g1, z4, z4, z8, 0);
#else
    __builtin_amdgcn_tensor_load_to_lds(g0, g1, z4, z4, 0);
#endif
}

// ---------------- fp32 -> bf16 weight/act cast ----------------
__global__ void cast_f32_bf16(const float* __restrict__ src, __bf16* __restrict__ dst, int n) {
    int i = blockIdx.x * blockDim.x + threadIdx.x;
    if (i < n) dst[i] = (__bf16)src[i];
}

// ---------------- RMSNorm + bf16 cast (one block per token row) ----------------
__global__ __launch_bounds__(256) void rmsnorm_cast(
    const float* __restrict__ x, const float* __restrict__ w, __bf16* __restrict__ out) {
    const int row = blockIdx.x;
    const float* xr = x + (size_t)row * D_MODEL;
    float s = 0.0f;
    for (int i = threadIdx.x; i < D_MODEL; i += 256) {
        float v = xr[i];
        s += v * v;
    }
    for (int off = 16; off > 0; off >>= 1) s += __shfl_down(s, off, 32);
    __shared__ float red[8];
    if ((threadIdx.x & 31) == 0) red[threadIdx.x >> 5] = s;
    __syncthreads();
    float tot = 0.0f;
#pragma unroll
    for (int i = 0; i < 8; ++i) tot += red[i];
    const float rinv = rsqrtf(tot / (float)D_MODEL + 1e-6f);
    for (int i = threadIdx.x; i < D_MODEL; i += 256)
        out[(size_t)row * D_MODEL + i] = (__bf16)(xr[i] * rinv * w[i]);
}

// ---------------- bf16 WMMA GEMM: C[M,N] = A[M,K] * W[N,K]^T ----------------
// block tile 128x128; tiles staged via double-buffered TDM DMA (BK=64 per stage,
// two 16x16x32 WMMA K-slabs). Wave0 issues the DMA for stage k+1 before computing
// stage k, so the tensor DMA overlaps the WMMA stream. 8 waves = 2(M) x 4(N).
// mode 0: bf16 store; mode 1: f32 store; mode 2: f32 store + residual add.
static constexpr int BM = 128, BN = 128, BKS = 64;

__global__ __launch_bounds__(256) void gemm_bf16_wmma(
    const __bf16* __restrict__ A, const __bf16* __restrict__ W,
    void* __restrict__ Cout, const float* __restrict__ residual,
    int M, int N, int K, int mode) {
    __shared__ __bf16 As[2][BM][BKS];
    __shared__ __bf16 Ws[2][BN][BKS];

    const int tid  = threadIdx.x;
    const int lane = tid & 31;
    const int wave = tid >> 5;
    const int wm   = wave >> 2;       // 0..1
    const int wn   = wave & 3;        // 0..3
    const int half = lane >> 4;       // hi/lo 16-lane group
    const int l16  = lane & 15;
    const int bm0  = blockIdx.y * BM;
    const int bn0  = blockIdx.x * BN;

    const unsigned wrows = (unsigned)((N - bn0 < BN) ? (N - bn0) : BN);
    const int nk = K / BKS;

    v8f acc[4][2];
#pragma unroll
    for (int mf = 0; mf < 4; ++mf)
#pragma unroll
        for (int nf = 0; nf < 2; ++nf)
#pragma unroll
            for (int r = 0; r < 8; ++r) acc[mf][nf][r] = 0.0f;

    // prologue: DMA stage 0
    if (wave == 0) {
        tdm_load_2d(lds_off_of(&As[0][0][0]), A + (size_t)bm0 * K, BM, BM, BKS, (unsigned)K, 1);
        tdm_load_2d(lds_off_of(&Ws[0][0][0]), W + (size_t)bn0 * K, wrows, BN, BKS, (unsigned)K, 1);
        __builtin_amdgcn_s_wait_tensorcnt(0);
    }
    __syncthreads();

    for (int ki = 0; ki < nk; ++ki) {
        const int buf = ki & 1;
        // issue DMA for next stage into the other buffer (overlaps compute below)
        if (wave == 0 && ki + 1 < nk) {
            const int k1 = (ki + 1) * BKS;
            tdm_load_2d(lds_off_of(&As[buf ^ 1][0][0]), A + (size_t)bm0 * K + k1,
                        BM, BM, BKS, (unsigned)K, 1);
            tdm_load_2d(lds_off_of(&Ws[buf ^ 1][0][0]), W + (size_t)bn0 * K + k1,
                        wrows, BN, BKS, (unsigned)K, 1);
        }

#pragma unroll
        for (int kk = 0; kk < BKS; kk += 32) {
            // fragment gathers per ISA 7.12.2 16-bit layouts
            v16bf afr[4], bfr[2];
#pragma unroll
            for (int mf = 0; mf < 4; ++mf) {
                int row = wm * 64 + mf * 16 + l16;
                afr[mf] = load_frag16(&As[buf][row][kk + half * 8],
                                      &As[buf][row][kk + 16 + half * 8]);
            }
#pragma unroll
            for (int nf = 0; nf < 2; ++nf) {
                int row = wn * 32 + nf * 16 + l16;
                bfr[nf] = load_frag16(&Ws[buf][row][kk + half * 8],
                                      &Ws[buf][row][kk + 16 + half * 8]);
            }
#pragma unroll
            for (int mf = 0; mf < 4; ++mf)
#pragma unroll
                for (int nf = 0; nf < 2; ++nf)
                    acc[mf][nf] = __builtin_amdgcn_wmma_f32_16x16x32_bf16(
                        false, afr[mf], false, bfr[nf], (short)0, acc[mf][nf], false, false);
        }

        if (wave == 0 && ki + 1 < nk) __builtin_amdgcn_s_wait_tensorcnt(0);
        __syncthreads();
    }

    // store: C/D layout -> element (r, lane): row = r + 8*half, col = l16
#pragma unroll
    for (int mf = 0; mf < 4; ++mf)
#pragma unroll
        for (int nf = 0; nf < 2; ++nf) {
            int col = bn0 + wn * 32 + nf * 16 + l16;
            if (col >= N) continue;
            int rbase = bm0 + wm * 64 + mf * 16 + half * 8;
#pragma unroll
            for (int r = 0; r < 8; ++r) {
                float v = acc[mf][nf][r];
                size_t idx = (size_t)(rbase + r) * N + col;
                if (mode == 0)      ((__bf16*)Cout)[idx] = (__bf16)v;
                else if (mode == 1) ((float*)Cout)[idx]  = v;
                else                ((float*)Cout)[idx]  = v + residual[idx];
            }
        }
}

// ---------------- causal depthwise conv (K=4) + bias + SiLU ----------------
__global__ void conv_silu(const __bf16* __restrict__ xz, const float* __restrict__ cw,
                          const float* __restrict__ cb, __bf16* __restrict__ xs) {
    int idx = blockIdx.x * blockDim.x + threadIdx.x;   // over MROWS * D_INNER
    if (idx >= MROWS * D_INNER) return;
    int c  = idx & (D_INNER - 1);
    int bt = idx >> 11;                                // D_INNER = 2048
    int t  = bt & (SEQ - 1);
    int b  = bt >> 11;                                 // SEQ = 2048
    float acc = cb[c];
#pragma unroll
    for (int j = 0; j < D_CONV; ++j) {
        int tt = t - (D_CONV - 1) + j;
        if (tt >= 0)
            acc += cw[c * D_CONV + j] *
                   (float)xz[((size_t)(b * SEQ + tt)) * N_XZ + c];
    }
    xs[(size_t)bt * D_INNER + c] = (__bf16)silu_f(acc);
}

// ---------------- selective scan, one block per (b,h); fused D-skip + gate ----------------
// B/C rows for 16 timesteps are staged per TDM DMA (strided 2D tile), double-buffered
// so the DMA for chunk c+1 overlaps the recurrence arithmetic of chunk c.
static constexpr int TCH = 16;                 // timesteps per staged chunk

__global__ __launch_bounds__(256) void ssm_scan(
    const float* __restrict__ bcdt, const __bf16* __restrict__ xs,
    const __bf16* __restrict__ xz, const float* __restrict__ A_log,
    const float* __restrict__ dt_b, const float* __restrict__ Dv,
    __bf16* __restrict__ yg) {
    const int b = blockIdx.x >> 3;
    const int h = blockIdx.x & 7;
    const int d = threadIdx.x;                 // 0..255 = D_HEAD
    const int wave = threadIdx.x >> 5;
    const float Ah  = -__expf(A_log[h]);
    const float dtb = dt_b[h];
    const float Dh  = Dv[h];

    float hs[D_STATE];
#pragma unroll
    for (int n = 0; n < D_STATE; ++n) hs[n] = 0.0f;

    __shared__ float sB[2][TCH][D_STATE];
    __shared__ float sC[2][TCH][D_STATE];

    const float* base = bcdt + (size_t)b * SEQ * N_BCDT;
    const int nch = SEQ / TCH;

    // prologue: stage chunk 0 (B block and C block, strided rows)
    if (wave == 0) {
        tdm_load_2d(lds_off_of(&sB[0][0][0]), base + h * D_STATE,
                    TCH, TCH, D_STATE, (unsigned)N_BCDT, 2);
        tdm_load_2d(lds_off_of(&sC[0][0][0]), base + 512 + h * D_STATE,
                    TCH, TCH, D_STATE, (unsigned)N_BCDT, 2);
        __builtin_amdgcn_s_wait_tensorcnt(0);
    }
    __syncthreads();

    for (int c = 0; c < nch; ++c) {
        const int buf = c & 1;
        if (wave == 0 && c + 1 < nch) {
            const float* nx = base + (size_t)(c + 1) * TCH * N_BCDT;
            tdm_load_2d(lds_off_of(&sB[buf ^ 1][0][0]), nx + h * D_STATE,
                        TCH, TCH, D_STATE, (unsigned)N_BCDT, 2);
            tdm_load_2d(lds_off_of(&sC[buf ^ 1][0][0]), nx + 512 + h * D_STATE,
                        TCH, TCH, D_STATE, (unsigned)N_BCDT, 2);
        }

        for (int tt = 0; tt < TCH; ++tt) {
            const int t = c * TCH + tt;
            const size_t row = (size_t)(b * SEQ + t);
            float dtr = bcdt[row * N_BCDT + 1024 + h] + dtb;
            float dt  = (dtr > 20.0f) ? dtr : log1pf(__expf(dtr));   // softplus
            float dA  = __expf(dt * Ah);
            float xt  = (float)xs[row * D_INNER + h * D_HEAD + d];
            float dtx = dt * xt;
            float y   = 0.0f;
#pragma unroll
            for (int n = 0; n < D_STATE; ++n) {
                hs[n] = dA * hs[n] + sB[buf][tt][n] * dtx;
                y    += sC[buf][tt][n] * hs[n];
            }
            y += xt * Dh;
            float z = (float)xz[row * N_XZ + D_INNER + h * D_HEAD + d];
            yg[row * D_INNER + h * D_HEAD + d] = (__bf16)(y * silu_f(z));
        }

        if (wave == 0 && c + 1 < nch) __builtin_amdgcn_s_wait_tensorcnt(0);
        __syncthreads();
    }
}

// ---------------- launch ----------------
extern "C" void kernel_launch(void* const* d_in, const int* in_sizes, int n_in,
                              void* d_out, int out_size, void* d_ws, size_t ws_size,
                              hipStream_t stream) {
    const float* x      = (const float*)d_in[0];
    const float* norm_w = (const float*)d_in[1];
    const float* in_w   = (const float*)d_in[2];
    const float* conv_w = (const float*)d_in[3];
    const float* conv_b = (const float*)d_in[4];
    const float* A_log  = (const float*)d_in[5];
    const float* B_w    = (const float*)d_in[6];
    const float* C_w    = (const float*)d_in[7];
    const float* dt_w   = (const float*)d_in[8];
    const float* dt_b   = (const float*)d_in[9];
    const float* Dv     = (const float*)d_in[10];
    const float* out_w  = (const float*)d_in[11];

    char* w = (char*)d_ws;
    __bf16* xn_b  = (__bf16*)w; w += (size_t)MROWS * D_MODEL * 2;
    __bf16* w_in  = (__bf16*)w; w += (size_t)N_XZ * D_MODEL * 2;
    __bf16* w_bc  = (__bf16*)w; w += (size_t)N_BCDT * D_INNER * 2;
    __bf16* w_out = (__bf16*)w; w += (size_t)D_MODEL * D_INNER * 2;
    __bf16* xz    = (__bf16*)w; w += (size_t)MROWS * N_XZ * 2;
    __bf16* xs    = (__bf16*)w; w += (size_t)MROWS * D_INNER * 2;
    float*  bcdt  = (float*) w; w += (size_t)MROWS * N_BCDT * 4;
    __bf16* yg    = (__bf16*)w; w += (size_t)MROWS * D_INNER * 2;

    auto blocks = [](int n) { return (n + 255) / 256; };

    // weight casts (fp32 -> bf16); B/C/dt packed into one [1032 x 2048] operand
    cast_f32_bf16<<<blocks(N_XZ * D_MODEL), 256, 0, stream>>>(in_w, w_in, N_XZ * D_MODEL);
    cast_f32_bf16<<<blocks(512 * D_INNER), 256, 0, stream>>>(B_w, w_bc, 512 * D_INNER);
    cast_f32_bf16<<<blocks(512 * D_INNER), 256, 0, stream>>>(C_w, w_bc + (size_t)512 * D_INNER, 512 * D_INNER);
    cast_f32_bf16<<<blocks(8 * D_INNER), 256, 0, stream>>>(dt_w, w_bc + (size_t)1024 * D_INNER, 8 * D_INNER);
    cast_f32_bf16<<<blocks(D_MODEL * D_INNER), 256, 0, stream>>>(out_w, w_out, D_MODEL * D_INNER);

    // 1) RMSNorm -> bf16
    rmsnorm_cast<<<MROWS, 256, 0, stream>>>(x, norm_w, xn_b);

    // 2) in_proj: [4096,1024] x [4096,1024]^T -> xz bf16 [4096,4096]
    gemm_bf16_wmma<<<dim3(N_XZ / BN, MROWS / BM), 256, 0, stream>>>(
        xn_b, w_in, xz, nullptr, MROWS, N_XZ, D_MODEL, 0);

    // 3) causal depthwise conv + SiLU -> xs bf16
    conv_silu<<<blocks(MROWS * D_INNER), 256, 0, stream>>>(xz, conv_w, conv_b, xs);

    // 4) fused B/C/dt projection: [4096,2048] x [1032,2048]^T -> f32
    gemm_bf16_wmma<<<dim3((N_BCDT + BN - 1) / BN, MROWS / BM), 256, 0, stream>>>(
        xs, w_bc, bcdt, nullptr, MROWS, N_BCDT, D_INNER, 1);

    // 5) selective scan + D-skip + SiLU(z) gate -> yg bf16
    ssm_scan<<<BATCH * N_HEADS, 256, 0, stream>>>(bcdt, xs, xz, A_log, dt_b, Dv, yg);

    // 6) out_proj + residual: [4096,2048] x [1024,2048]^T + x -> d_out f32
    gemm_bf16_wmma<<<dim3(D_MODEL / BN, MROWS / BM), 256, 0, stream>>>(
        yg, w_out, (float*)d_out, x, MROWS, D_MODEL, D_INNER, 2);
}
